// Seq2Seq_39238821216391
// MI455X (gfx1250) — matrix operand order repaired and checked by reading, not compiled, not run on hardware
//
#include <hip/hip_runtime.h>
#include <hip/hip_bf16.h>
#include <math.h>

typedef _Float16 f16;
typedef __attribute__((ext_vector_type(16))) _Float16 v16h;
typedef __attribute__((ext_vector_type(8)))  _Float16 v8h;
typedef __attribute__((ext_vector_type(8)))  float    v8f;

#define WMMA_F32_F16(a,b,c) \
  __builtin_amdgcn_wmma_f32_16x16x32_f16(false,(a),false,(b),(short)0,(c),false,false)

// Build the 16x32 f16 A-fragment for one lane in contiguous k order (B loads use the same):
// lane l (l<16): row M=l, K = {kb..kb+15} with kb=k0
// lane l (l>=16): same rows, kb=k0+16. Two contiguous 16-byte chunks.
static __device__ inline v16h make_a(const f16* rowp) {
  v8h lo = *(const v8h*)rowp;        // K = kb .. kb+7
  v8h hh = *(const v8h*)(rowp + 8);  // K = kb+8 .. kb+15
  v16h a;
#pragma unroll
  for (int i = 0; i < 8; i++) { a[i] = lo[i]; a[i + 8] = hh[i]; }
  return a;
}

static __device__ inline float sigf(float x) { return 1.0f / (1.0f + __expf(-x)); }

// ---------------- f32 -> f16 convert ----------------
__global__ void k_cvt_f16(const float* __restrict__ s, f16* __restrict__ d, int n) {
  int i = blockIdx.x * blockDim.x + threadIdx.x;
  if (i < n) d[i] = (f16)s[i];
}

// ---------------- embedding gather (time-major rows: row = t*32 + b) ----------
__global__ void k_embed(const int* __restrict__ tok, const float* __restrict__ emb,
                        f16* __restrict__ X) {
  int row = blockIdx.x;            // rows = T*32
  int t = row >> 5, b = row & 31;
  int id = tok[b * 64 + t];        // tgt[:, :-1] handled by launching T=63 rows
  const float* e = emb + (size_t)id * 256;
  f16* x = X + (size_t)row * 256;
  for (int i = threadIdx.x; i < 256; i += blockDim.x) x[i] = (f16)e[i];
}

// ---------------- WMMA GEMM, 2M x 2N register blocking, SW-pipelined ----------
// C[Rows,N] = A[Rows,K]f16 @ W[N,K]f16^T + bias1 + bias2.
// Iteration k+1's 8 b128 loads are issued before iteration k's 4 v_wmma, so
// WMMAs only wait on loads with a full iteration of latency cover. 2Mx2N keeps
// live VGPRs (~96 + addressing) spill-free.
__global__ void k_gemm_bias(const f16* __restrict__ A, const f16* __restrict__ W,
                            const float* __restrict__ bias1,
                            const float* __restrict__ bias2,
                            float* __restrict__ C, int Rows, int K, int N) {
  int wave = blockIdx.x * (blockDim.x >> 5) + (threadIdx.x >> 5);
  int lane = threadIdx.x & 31;
  int lane15 = lane & 15, hi = (lane >> 4) & 1;
  int Ng = N >> 5;        // groups of 2 n-tiles
  int Mg = Rows >> 5;     // groups of 2 m-tiles
  if (wave >= Mg * Ng) return;
  int mg = wave / Ng, ng = wave - mg * Ng;
  int m0 = mg << 5, n0 = ng << 5;
  v8f acc00 = {}, acc01 = {}, acc10 = {}, acc11 = {};
  const f16* arow0 = A + (size_t)(m0 + lane15) * K + (hi ? 16 : 0);
  const f16* arow1 = arow0 + (size_t)16 * K;
  const f16* wb = W + (size_t)(n0 + lane15) * K + (hi ? 16 : 0); // B[k][n]=W[n][k]
  const size_t cs = (size_t)16 * K;                              // n-tile stride

  v16h a0 = make_a(arow0), a1 = make_a(arow1);
  v16h b0 = *(const v16h*)(wb);
  v16h b1 = *(const v16h*)(wb + cs);
#pragma unroll 2
  for (int k0 = 0; k0 < K - 32; k0 += 32) {
    int kn = k0 + 32;
    v16h na0 = make_a(arow0 + kn);
    v16h na1 = make_a(arow1 + kn);
    v16h nb0 = *(const v16h*)(wb + kn);
    v16h nb1 = *(const v16h*)(wb + kn + cs);
    acc00 = WMMA_F32_F16(a0, b0, acc00);
    acc10 = WMMA_F32_F16(a1, b0, acc10);
    acc01 = WMMA_F32_F16(a0, b1, acc01);
    acc11 = WMMA_F32_F16(a1, b1, acc11);
    a0 = na0; a1 = na1; b0 = nb0; b1 = nb1;
  }
  acc00 = WMMA_F32_F16(a0, b0, acc00);
  acc10 = WMMA_F32_F16(a1, b0, acc10);
  acc01 = WMMA_F32_F16(a0, b1, acc01);
  acc11 = WMMA_F32_F16(a1, b1, acc11);

  v8f accs[2][2] = {{acc00, acc01}, {acc10, acc11}};
#pragma unroll
  for (int j = 0; j < 2; j++) {
    int col = n0 + j * 16 + lane15;
    float bias = 0.f;
    if (bias1) bias += bias1[col];
    if (bias2) bias += bias2[col];
#pragma unroll
    for (int mi = 0; mi < 2; mi++) {
      int rbase = m0 + mi * 16 + hi * 8;
#pragma unroll
      for (int r = 0; r < 8; r++)
        C[(size_t)(rbase + r) * N + col] = accs[mi][j][r] + bias;
    }
  }
}

// ---------------- output projection: logits[b, t+1, v], 2M x 2N, pipelined -----
__global__ void k_gemm_logits(const f16* __restrict__ A,   // [2016, 512] time-major
                              const f16* __restrict__ W,   // [16000, 512]
                              const float* __restrict__ bias,
                              float* __restrict__ out) {   // [32, 64, 16000]
  const int K = 512;
  const int Mg = 63, Ng = 500;         // 126 m-tiles / 2, 1000 n-tiles / 2
  int wave = blockIdx.x * (blockDim.x >> 5) + (threadIdx.x >> 5);
  int lane = threadIdx.x & 31, lane15 = lane & 15, hi = (lane >> 4) & 1;
  if (wave >= Mg * Ng) return;
  int mg = wave / Ng, ng = wave - mg * Ng;
  int m0 = mg << 5, n0 = ng << 5;
  v8f acc00 = {}, acc01 = {}, acc10 = {}, acc11 = {};
  const f16* arow0 = A + (size_t)(m0 + lane15) * K + (hi ? 16 : 0);
  const f16* arow1 = arow0 + (size_t)16 * K;
  const f16* wb = W + (size_t)(n0 + lane15) * K + (hi ? 16 : 0);
  const size_t cs = (size_t)16 * K;

  v16h a0 = make_a(arow0), a1 = make_a(arow1);
  v16h b0 = *(const v16h*)(wb);
  v16h b1 = *(const v16h*)(wb + cs);
#pragma unroll 2
  for (int k0 = 0; k0 < K - 32; k0 += 32) {
    int kn = k0 + 32;
    v16h na0 = make_a(arow0 + kn);
    v16h na1 = make_a(arow1 + kn);
    v16h nb0 = *(const v16h*)(wb + kn);
    v16h nb1 = *(const v16h*)(wb + kn + cs);
    acc00 = WMMA_F32_F16(a0, b0, acc00);
    acc10 = WMMA_F32_F16(a1, b0, acc10);
    acc01 = WMMA_F32_F16(a0, b1, acc01);
    acc11 = WMMA_F32_F16(a1, b1, acc11);
    a0 = na0; a1 = na1; b0 = nb0; b1 = nb1;
  }
  acc00 = WMMA_F32_F16(a0, b0, acc00);
  acc10 = WMMA_F32_F16(a1, b0, acc10);
  acc01 = WMMA_F32_F16(a0, b1, acc01);
  acc11 = WMMA_F32_F16(a1, b1, acc11);

  v8f accs[2][2] = {{acc00, acc01}, {acc10, acc11}};
#pragma unroll
  for (int j = 0; j < 2; j++) {
    int col = n0 + j * 16 + lane15;
    float bv = bias[col];
#pragma unroll
    for (int mi = 0; mi < 2; mi++) {
      int rbase = m0 + mi * 16 + hi * 8;
#pragma unroll
      for (int r = 0; r < 8; r++) {
        int row = rbase + r;           // row = t*32 + b
        int t = row >> 5, b = row & 31;
        out[((size_t)b * 64 + (t + 1)) * 16000 + col] = accs[mi][j][r] + bv;
      }
    }
  }
}

__global__ void k_zero_row0(float* __restrict__ out) {
  int i = blockIdx.x * blockDim.x + threadIdx.x;
  if (i < 32 * 16000) {
    int b = i / 16000, v = i - b * 16000;
    out[(size_t)b * 64 * 16000 + v] = 0.f;
  }
}

// ---------------- persistent recurrent LSTM: one WGP, 32 waves -----------------
// gates[32,2048] = h[32,512] @ Whh^T + Gin[t];  i,f,g,o blocks of 512.
// Each wave owns 2 (m,n) positions; per position the A fragment (h from LDS) is
// shared across the 4 gate WMMAs, with the 4 global B streams SW-pipelined one
// k-step ahead. EXEC all-ones around every WMMA.
__global__ void __launch_bounds__(1024, 1)
k_lstm_seq(const f16* __restrict__ Whh,   // f16 [2048, 512]
           const float* __restrict__ Gin, // [T*32, 2048], bias already folded in
           const float* __restrict__ h0, const float* __restrict__ c0,
           f16* __restrict__ Yout,        // f16 [T*32, 512]
           float* __restrict__ hT, float* __restrict__ cT, int T) {
  __shared__ f16 h16[32 * 512];          // 32 KB: current h in f16 (WMMA A operand)
  const int tid = threadIdx.x;
  const int wave = tid >> 5, lane = tid & 31;
  const int lane15 = lane & 15, hi = (lane >> 4) & 1;
  const int GS = 512 * 512;              // gate stride in Whh (halfs)

  for (int i = tid; i < 32 * 512; i += 1024) h16[i] = (f16)(h0 ? h0[i] : 0.0f);

  float creg[2][8];                      // cell state lives in registers
  float hreg[2][8] = {};
#pragma unroll
  for (int q = 0; q < 2; q++) {
    int p = wave * 2 + q;
    int m0 = (p >> 5) << 4, n0 = (p & 31) << 4;
    int col = n0 + lane15, rbase = m0 + hi * 8;
#pragma unroll
    for (int r = 0; r < 8; r++)
      creg[q][r] = c0 ? c0[(rbase + r) * 512 + col] : 0.0f;
  }
  __syncthreads();

  for (int t = 0; t < T; t++) {
    const float* gin = Gin + (size_t)t * (32 * 2048);
#pragma unroll
    for (int q = 0; q < 2; q++) {
      int p = wave * 2 + q;
      int m0 = (p >> 5) << 4, n0 = (p & 31) << 4;
      v8f c0a = {}, c1a = {}, c2a = {}, c3a = {};
      const f16* wb  = Whh + (size_t)(n0 + lane15) * 512 + (hi ? 16 : 0);
      const f16* ap0 = &h16[(m0 + lane15) * 512 + (hi ? 16 : 0)];

      v16h a  = make_a(ap0);                         // ds_load_b128 x2
      v16h b0 = *(const v16h*)(wb);
      v16h b1 = *(const v16h*)(wb + GS);
      v16h b2 = *(const v16h*)(wb + 2 * GS);
      v16h b3 = *(const v16h*)(wb + 3 * GS);
#pragma unroll 2
      for (int k0 = 0; k0 < 512 - 32; k0 += 32) {
        int kn = k0 + 32;
        v16h na  = make_a(ap0 + kn);
        v16h nb0 = *(const v16h*)(wb + kn);
        v16h nb1 = *(const v16h*)(wb + kn + GS);
        v16h nb2 = *(const v16h*)(wb + kn + 2 * GS);
        v16h nb3 = *(const v16h*)(wb + kn + 3 * GS);
        c0a = WMMA_F32_F16(a, b0, c0a);
        c1a = WMMA_F32_F16(a, b1, c1a);
        c2a = WMMA_F32_F16(a, b2, c2a);
        c3a = WMMA_F32_F16(a, b3, c3a);
        a = na; b0 = nb0; b1 = nb1; b2 = nb2; b3 = nb3;
      }
      c0a = WMMA_F32_F16(a, b0, c0a);
      c1a = WMMA_F32_F16(a, b1, c1a);
      c2a = WMMA_F32_F16(a, b2, c2a);
      c3a = WMMA_F32_F16(a, b3, c3a);

      int col = n0 + lane15, rbase = m0 + hi * 8;
#pragma unroll
      for (int r = 0; r < 8; r++) {
        int brow = rbase + r;
        const float* g = gin + (size_t)brow * 2048 + col;
        float iv = c0a[r] + g[0];
        float fv = c1a[r] + g[512];
        float gv = c2a[r] + g[1024];
        float ov = c3a[r] + g[1536];
        float cn = sigf(fv) * creg[q][r] + sigf(iv) * tanhf(gv);
        float hn = sigf(ov) * tanhf(cn);
        creg[q][r] = cn;
        hreg[q][r] = hn;
        Yout[((size_t)t * 32 + brow) * 512 + col] = (f16)hn;
      }
    }
    __syncthreads();                     // all waves done reading old h16
#pragma unroll
    for (int q = 0; q < 2; q++) {
      int p = wave * 2 + q;
      int m0 = (p >> 5) << 4, n0 = (p & 31) << 4;
      int col = n0 + lane15, rbase = m0 + hi * 8;
#pragma unroll
      for (int r = 0; r < 8; r++) h16[(rbase + r) * 512 + col] = (f16)hreg[q][r];
    }
    __syncthreads();                     // publish new h for next step
  }

  if (hT && cT) {
#pragma unroll
    for (int q = 0; q < 2; q++) {
      int p = wave * 2 + q;
      int m0 = (p >> 5) << 4, n0 = (p & 31) << 4;
      int col = n0 + lane15, rbase = m0 + hi * 8;
#pragma unroll
      for (int r = 0; r < 8; r++) {
        hT[(rbase + r) * 512 + col] = hreg[q][r];
        cT[(rbase + r) * 512 + col] = creg[q][r];
      }
    }
  }
}

// ------------------------------ host side --------------------------------------
extern "C" void kernel_launch(void* const* d_in, const int* in_sizes, int n_in,
                              void* d_out, int out_size, void* d_ws, size_t ws_size,
                              hipStream_t stream) {
  (void)in_sizes; (void)n_in; (void)out_size;
  const int*   src     = (const int*)d_in[0];
  const int*   tgt     = (const int*)d_in[1];
  const float* enc_emb = (const float*)d_in[2];
  const float* encW[2][4];
  for (int l = 0; l < 2; l++) for (int j = 0; j < 4; j++) encW[l][j] = (const float*)d_in[3 + l * 4 + j];
  const float* dec_emb = (const float*)d_in[11];
  const float* decW[3][4];
  for (int l = 0; l < 3; l++) for (int j = 0; j < 4; j++) decW[l][j] = (const float*)d_in[12 + l * 4 + j];
  const float* W_out = (const float*)d_in[24];
  const float* b_out = (const float*)d_in[25];
  float* out = (float*)d_out;

  char* ws = (char*)d_ws;
  size_t cur = 0;
  auto alloc = [&](size_t bytes) -> char* {
    char* p = ws + cur;
    cur += (bytes + 255) & ~(size_t)255;
    return p;
  };

  // layer order: enc0, enc1, dec0, dec1, dec2
  const int Din[5] = {256, 512, 256, 512, 512};
  f16* wWih[5]; f16* wWhh[5];
  for (int l = 0; l < 5; l++) {
    wWih[l] = (f16*)alloc((size_t)2048 * Din[l] * 2);
    wWhh[l] = (f16*)alloc((size_t)2048 * 512 * 2);
  }
  f16*   wWout = (f16*)alloc((size_t)16000 * 512 * 2);
  f16*   Xe    = (f16*)alloc((size_t)2048 * 256 * 2);
  f16*   Xd    = (f16*)alloc((size_t)2016 * 256 * 2);
  float* Gin   = (float*)alloc((size_t)2048 * 2048 * 4);
  f16*   Ya    = (f16*)alloc((size_t)2048 * 512 * 2);
  f16*   Yb    = (f16*)alloc((size_t)2048 * 512 * 2);
  float* encH  = (float*)alloc((size_t)2 * 32 * 512 * 4);
  float* encC  = (float*)alloc((size_t)2 * 32 * 512 * 4);
  float* dumpH = (float*)alloc((size_t)32 * 512 * 4);
  float* dumpC = (float*)alloc((size_t)32 * 512 * 4);
  if (cur > ws_size) return;  // workspace too small; bail deterministically

  // 1) weight conversions f32 -> f16
  const float* cs[11] = {encW[0][0], encW[0][1], encW[1][0], encW[1][1],
                         decW[0][0], decW[0][1], decW[1][0], decW[1][1],
                         decW[2][0], decW[2][1], W_out};
  f16* cd[11] = {wWih[0], wWhh[0], wWih[1], wWhh[1], wWih[2], wWhh[2],
                 wWih[3], wWhh[3], wWih[4], wWhh[4], wWout};
  const int cn[11] = {2048 * 256, 2048 * 512, 2048 * 512, 2048 * 512,
                      2048 * 256, 2048 * 512, 2048 * 512, 2048 * 512,
                      2048 * 512, 2048 * 512, 16000 * 512};
  for (int i = 0; i < 11; i++)
    k_cvt_f16<<<(cn[i] + 255) / 256, 256, 0, stream>>>(cs[i], cd[i], cn[i]);

  // 2) embedding gathers (time-major)
  k_embed<<<2048, 256, 0, stream>>>(src, enc_emb, Xe);  // 64*32 rows
  k_embed<<<2016, 256, 0, stream>>>(tgt, dec_emb, Xd);  // 63*32 rows (tgt[:, :-1])

  // 3) LSTM layers: big parallel input GEMM, then persistent recurrent kernel
  auto run_layer = [&](const f16* X, int rows, int layer,
                       const float* h0, const float* c0,
                       f16* Y, float* hTo, float* cTo, int T) {
    int waves = (rows >> 5) * (2048 >> 5);   // 2Mx2N blocks
    k_gemm_bias<<<waves / 8, 256, 0, stream>>>(X, wWih[layer],
                                               layer < 2 ? encW[layer][2] : decW[layer - 2][2],
                                               layer < 2 ? encW[layer][3] : decW[layer - 2][3],
                                               Gin, rows, Din[layer], 2048);
    k_lstm_seq<<<1, 1024, 0, stream>>>(wWhh[layer], Gin, h0, c0, Y, hTo, cTo, T);
  };
  run_layer(Xe, 2048, 0, nullptr, nullptr, Ya, encH,          encC,          64);
  run_layer(Ya, 2048, 1, nullptr, nullptr, Yb, encH + 16384,  encC + 16384,  64);
  run_layer(Xd, 2016, 2, encH,         encC,         Ya, dumpH, dumpC, 63);
  run_layer(Ya, 2016, 3, encH + 16384, encC + 16384, Yb, dumpH, dumpC, 63);
  run_layer(Yb, 2016, 4, encH + 16384, encC + 16384, Ya, dumpH, dumpC, 63);

  // 4) logits: zero t=0 row, then big WMMA projection
  k_zero_row0<<<(32 * 16000 + 255) / 256, 256, 0, stream>>>(out);
  k_gemm_logits<<<(63 * 500 + 7) / 8, 256, 0, stream>>>(Ya, wWout, b_out, out);
}
